// Unet_32212254720644
// MI455X (gfx1250) — matrix operand order repaired and checked by reading, not compiled, not run on hardware
//
#include <hip/hip_runtime.h>
#include <cstdint>
#include <cstddef>

// ---------------- model constants ----------------
#define NN 4096          // nodes
#define FD 256           // feature dim
static const int KK0 = 3276;     // int(0.8*4096)
static const int KK0_LD = 3280;  // padded ld: bf16 rows 16B-aligned
static const int KK1 = 1965;     // int(0.6*3276)
static const int KK1_LD = 1968;  // padded ld: rows 16B-aligned

typedef __attribute__((ext_vector_type(16))) __bf16 v16bf;
typedef __attribute__((ext_vector_type(8)))  float  v8f;

__device__ __forceinline__ unsigned short f2bfu(float f) {
  unsigned u = __builtin_bit_cast(unsigned, f);
  u = (u + 0x7FFFu + ((u >> 16) & 1u)) >> 16;          // RNE
  return (unsigned short)u;
}
__device__ __forceinline__ __bf16 f2bf(float f) {
  unsigned short s = f2bfu(f);
  return __builtin_bit_cast(__bf16, s);
}
__device__ __forceinline__ unsigned pk2(float lo, float hi) {
  return (unsigned)f2bfu(lo) | ((unsigned)f2bfu(hi) << 16);
}

// =====================================================================
// WMMA GEMM with bf16 A-matrix in memory: C[M,N] = A(bf16)[M,K] @ B(f32)[K,N]
// Double-buffered pipeline: tile t+1's A goes global->LDS via async b128
// copies (ASYNCcnt) while tile t's WMMAs run; one barrier per K-tile.
// =====================================================================
__global__ __launch_bounds__(256) void k_gemm_bfA(
    const __bf16* __restrict__ A, const float* __restrict__ B,
    float* __restrict__ C, int M, int N, int K, int lda, int ldb, int ldc)
{
  __shared__ __bf16 As[2][64 * 32];   // [m][k]
  __shared__ __bf16 Bs[2][64 * 32];   // [n][k]
  const int tid  = threadIdx.x;
  const int lane = tid & 31;
  const int wave = tid >> 5;
  const int m0 = blockIdx.y * 64;
  const int n0 = blockIdx.x * 64;
  const int tr  = wave & 3;
  const int tcb = (wave >> 2) * 2;
  const int half = lane >> 4;
  const int l16  = lane & 15;
  const bool interior = (m0 + 64 <= M) && (n0 + 64 <= N);
  const int rA = tid >> 2, cA = (tid & 3) << 3;   // A: 8 bf16 = 16B per thread
  const int kB = tid >> 3, nB = (tid & 7) << 3;   // B: 8 contiguous n (coalesced)

  v8f acc0 = {}; v8f acc1 = {};

  auto stage_fast = [&](int k0, int p) {
    // async A-tile: 16B global -> LDS per lane, no VGPR round trip
    const __bf16* srcA = A + (size_t)(m0 + rA) * lda + (k0 + cA);
    unsigned ldsA = (unsigned)(size_t)&As[p][rA * 32 + cA];
    unsigned long long ga = (unsigned long long)(size_t)srcA;
    asm volatile("global_load_async_to_lds_b128 %0, %1, off"
                 :: "v"(ldsA), "v"(ga) : "memory");
    // B-tile: vectorized f32 loads, pack to bf16 [n][k]
    const float* srcB = B + (size_t)(k0 + kB) * ldb + (n0 + nB);
    float4 f0 = *reinterpret_cast<const float4*>(srcB);
    float4 f1 = *reinterpret_cast<const float4*>(srcB + 4);
    __bf16* bs = &Bs[p][0];
    bs[(nB + 0) * 32 + kB] = f2bf(f0.x);
    bs[(nB + 1) * 32 + kB] = f2bf(f0.y);
    bs[(nB + 2) * 32 + kB] = f2bf(f0.z);
    bs[(nB + 3) * 32 + kB] = f2bf(f0.w);
    bs[(nB + 4) * 32 + kB] = f2bf(f1.x);
    bs[(nB + 5) * 32 + kB] = f2bf(f1.y);
    bs[(nB + 6) * 32 + kB] = f2bf(f1.z);
    bs[(nB + 7) * 32 + kB] = f2bf(f1.w);
  };

  auto stage_slow = [&](int k0, int p) {
    unsigned short z = 0;
#pragma unroll
    for (int i = 0; i < 8; ++i) {
      int idx = tid * 8 + i;
      int r = idx >> 5, c = idx & 31;
      int gm = m0 + r, gk = k0 + c;
      As[p][r * 32 + c] = (gm < M && gk < K) ? A[(size_t)gm * lda + gk]
                                             : __builtin_bit_cast(__bf16, z);
    }
#pragma unroll
    for (int i = 0; i < 8; ++i) {
      int idx = tid * 8 + i;
      int n = idx >> 5, c = idx & 31;
      int gn = n0 + n, gk = k0 + c;
      float v = (gn < N && gk < K) ? B[(size_t)gk * ldb + gn] : 0.f;
      Bs[p][n * 32 + c] = f2bf(v);
    }
  };

  auto stage_any = [&](int t, int p) {
    int k0 = t * 32;
    if (interior && (k0 + 32 <= K)) stage_fast(k0, p); else stage_slow(k0, p);
  };

  auto mma = [&](int p) {
    v16bf af;
    const __bf16* Ap = &As[p][(tr * 16 + l16) * 32];
    int ka = 16 * half, kb = 16 * half + 8;
#pragma unroll
    for (int e = 0; e < 8; ++e) { af[e] = Ap[ka + e]; af[e + 8] = Ap[kb + e]; }
    v16bf b0v, b1v;
    const __bf16* Bp0 = &Bs[p][(tcb * 16 + l16) * 32 + 16 * half];
    const __bf16* Bp1 = &Bs[p][((tcb + 1) * 16 + l16) * 32 + 16 * half];
#pragma unroll
    for (int e = 0; e < 16; ++e) { b0v[e] = Bp0[e]; b1v[e] = Bp1[e]; }
    acc0 = __builtin_amdgcn_wmma_f32_16x16x32_bf16(false, af, false, b0v,
                                                   (short)0, acc0, false, false);
    acc1 = __builtin_amdgcn_wmma_f32_16x16x32_bf16(false, af, false, b1v,
                                                   (short)0, acc1, false, false);
  };

  const int nt = (K + 31) / 32;
  stage_any(0, 0);
  asm volatile("s_wait_asynccnt 0x0" ::: "memory");
  __syncthreads();
  for (int t = 0; t < nt; ++t) {
    int p = t & 1;
    if (t + 1 < nt) stage_any(t + 1, p ^ 1);   // overlaps with WMMAs below
    mma(p);
    asm volatile("s_wait_asynccnt 0x0" ::: "memory");  // our async landed
    __syncthreads();                                    // everyone's landed
  }

#pragma unroll
  for (int v = 0; v < 8; ++v) {
    int m = m0 + tr * 16 + v + 8 * half;
    if (m >= M) continue;
#pragma unroll
    for (int t = 0; t < 2; ++t) {
      int n = n0 + (tcb + t) * 16 + l16;
      if (n >= N) continue;
      C[(size_t)m * ldc + n] = (t == 0) ? acc0[v] : acc1[v];
    }
  }
}

// =====================================================================
// fp32-input bf16 WMMA GEMM (projections, reduce, final h1@h2^T)
// mode: 0 = none, 1 = bias+relu, 2 = bias, 3 = sigmoid ; transB: B is [N,K]
// =====================================================================
__global__ __launch_bounds__(256) void k_gemm_bf16(
    const float* __restrict__ A, const float* __restrict__ B,
    float* __restrict__ C, int M, int N, int K,
    int lda, int ldb, int ldc,
    const float* __restrict__ bias, int mode, int transB)
{
  __shared__ __bf16 As[64 * 32];   // [m][k]
  __shared__ __bf16 Bs[64 * 32];   // [n][k]
  const int tid  = threadIdx.x;
  const int lane = tid & 31;
  const int wave = tid >> 5;
  const int m0 = blockIdx.y * 64;
  const int n0 = blockIdx.x * 64;
  const int tr  = wave & 3;
  const int tcb = (wave >> 2) * 2;
  const int half = lane >> 4;
  const int l16  = lane & 15;

  const bool fullM = (m0 + 64 <= M);
  const bool fullN = (n0 + 64 <= N);

  v8f acc0 = {}; v8f acc1 = {};

  for (int k0 = 0; k0 < K; k0 += 32) {
    const bool fullK = (k0 + 32 <= K);

    if (fullM && fullK) {
      int r = tid >> 2;
      int c = (tid & 3) << 3;
      const float* src = A + (size_t)(m0 + r) * lda + (k0 + c);
      float4 f0 = *reinterpret_cast<const float4*>(src);
      float4 f1 = *reinterpret_cast<const float4*>(src + 4);
      uint4 p;
      p.x = pk2(f0.x, f0.y); p.y = pk2(f0.z, f0.w);
      p.z = pk2(f1.x, f1.y); p.w = pk2(f1.z, f1.w);
      *reinterpret_cast<uint4*>(&As[r * 32 + c]) = p;
    } else {
#pragma unroll
      for (int i = 0; i < 8; ++i) {
        int idx = tid * 8 + i;
        int r = idx >> 5, c = idx & 31;
        int gm = m0 + r, gk = k0 + c;
        float v = (gm < M && gk < K) ? A[(size_t)gm * lda + gk] : 0.f;
        As[r * 32 + c] = f2bf(v);
      }
    }

    if (transB) {
      if (fullN && fullK) {
        int r = tid >> 2;
        int c = (tid & 3) << 3;
        const float* src = B + (size_t)(n0 + r) * ldb + (k0 + c);
        float4 f0 = *reinterpret_cast<const float4*>(src);
        float4 f1 = *reinterpret_cast<const float4*>(src + 4);
        uint4 p;
        p.x = pk2(f0.x, f0.y); p.y = pk2(f0.z, f0.w);
        p.z = pk2(f1.x, f1.y); p.w = pk2(f1.z, f1.w);
        *reinterpret_cast<uint4*>(&Bs[r * 32 + c]) = p;
      } else {
#pragma unroll
        for (int i = 0; i < 8; ++i) {
          int idx = tid * 8 + i;
          int n = idx >> 5, c = idx & 31;
          int gn = n0 + n, gk = k0 + c;
          float v = (gn < N && gk < K) ? B[(size_t)gn * ldb + gk] : 0.f;
          Bs[n * 32 + c] = f2bf(v);
        }
      }
    } else {
      if (fullN && fullK) {
        int k = tid >> 3;
        int n = (tid & 7) << 3;
        const float* src = B + (size_t)(k0 + k) * ldb + (n0 + n);
        float4 f0 = *reinterpret_cast<const float4*>(src);
        float4 f1 = *reinterpret_cast<const float4*>(src + 4);
        Bs[(n + 0) * 32 + k] = f2bf(f0.x);
        Bs[(n + 1) * 32 + k] = f2bf(f0.y);
        Bs[(n + 2) * 32 + k] = f2bf(f0.z);
        Bs[(n + 3) * 32 + k] = f2bf(f0.w);
        Bs[(n + 4) * 32 + k] = f2bf(f1.x);
        Bs[(n + 5) * 32 + k] = f2bf(f1.y);
        Bs[(n + 6) * 32 + k] = f2bf(f1.z);
        Bs[(n + 7) * 32 + k] = f2bf(f1.w);
      } else {
#pragma unroll
        for (int i = 0; i < 8; ++i) {
          int idx = tid * 8 + i;
          int n = idx >> 5, c = idx & 31;
          int gn = n0 + n, gk = k0 + c;
          float v = (gn < N && gk < K) ? B[(size_t)gk * ldb + gn] : 0.f;
          Bs[n * 32 + c] = f2bf(v);
        }
      }
    }
    __syncthreads();

    v16bf af;
    {
      const __bf16* Ap = &As[(tr * 16 + l16) * 32];
      int ka = 16 * half, kb = 16 * half + 8;
#pragma unroll
      for (int e = 0; e < 8; ++e) { af[e] = Ap[ka + e]; af[e + 8] = Ap[kb + e]; }
    }
    v16bf bf0, bf1;
    {
      const __bf16* Bp0 = &Bs[(tcb * 16 + l16) * 32 + 16 * half];
      const __bf16* Bp1 = &Bs[((tcb + 1) * 16 + l16) * 32 + 16 * half];
#pragma unroll
      for (int e = 0; e < 16; ++e) { bf0[e] = Bp0[e]; bf1[e] = Bp1[e]; }
    }
    acc0 = __builtin_amdgcn_wmma_f32_16x16x32_bf16(false, af, false, bf0,
                                                   (short)0, acc0, false, false);
    acc1 = __builtin_amdgcn_wmma_f32_16x16x32_bf16(false, af, false, bf1,
                                                   (short)0, acc1, false, false);
    __syncthreads();
  }

#pragma unroll
  for (int v = 0; v < 8; ++v) {
    int m = m0 + tr * 16 + v + 8 * half;
    if (m >= M) continue;
#pragma unroll
    for (int t = 0; t < 2; ++t) {
      int n = n0 + (tcb + t) * 16 + l16;
      if (n >= N) continue;
      float x = (t == 0) ? acc0[v] : acc1[v];
      if (mode == 1)      { x += bias[n]; x = fmaxf(x, 0.f); }
      else if (mode == 2) { x += bias[n]; }
      else if (mode == 3) { x = 1.f / (1.f + expf(-x)); }
      C[(size_t)m * ldc + n] = x;
    }
  }
}

// ---------------- norm_g helpers ----------------
__global__ void k_rowdeg(const float* __restrict__ g, int cols, int ld,
                         float* __restrict__ deg) {
  __shared__ float red[256];
  int row = blockIdx.x, t = threadIdx.x;
  float s = 0.f;
  for (int j = t; j < cols; j += 256) s += g[(size_t)row * ld + j];
  red[t] = s; __syncthreads();
  for (int o = 128; o > 0; o >>= 1) { if (t < o) red[t] += red[t + o]; __syncthreads(); }
  if (t == 0) deg[row] = (red[0] == 0.f) ? 1.f : red[0];
}

// dst = src / deg[col]; optionally also writes a bf16 shadow copy (fused cvt)
__global__ void k_colscale(const float* __restrict__ src, float* __restrict__ dst,
                           __bf16* __restrict__ dstb, const float* __restrict__ deg,
                           size_t rows, size_t cols, size_t ld) {
  size_t total = rows * cols;
  size_t stride = (size_t)gridDim.x * blockDim.x;
  for (size_t i = (size_t)blockIdx.x * blockDim.x + threadIdx.x; i < total; i += stride) {
    size_t r = i / cols, c = i - r * cols;
    float v = src[r * ld + c] / deg[c];
    dst[r * ld + c] = v;
    if (dstb) dstb[r * ld + c] = f2bf(v);
  }
}

// ---------------- LayerNorm (row of 256): out = LN(a [+ b]) * s + bi --------
__global__ void k_layernorm(const float* __restrict__ a, const float* __restrict__ b,
                            const float* __restrict__ s, const float* __restrict__ bi,
                            float* __restrict__ out) {
  __shared__ float red[256];
  int row = blockIdx.x, t = threadIdx.x;
  float x = a[(size_t)row * FD + t];
  if (b) x += b[(size_t)row * FD + t];
  red[t] = x; __syncthreads();
  for (int o = 128; o > 0; o >>= 1) { if (t < o) red[t] += red[t + o]; __syncthreads(); }
  float mu = red[0] / 256.f; __syncthreads();
  float d = x - mu;
  red[t] = d * d; __syncthreads();
  for (int o = 128; o > 0; o >>= 1) { if (t < o) red[t] += red[t + o]; __syncthreads(); }
  float var = red[0] / 256.f;
  out[(size_t)row * FD + t] = d * rsqrtf(var + 1e-5f) * s[t] + bi[t];
}

// ---------------- pooling scores: sigmoid(h @ w + b) ----------------
__global__ void k_scores(const float* __restrict__ h, const float* __restrict__ w,
                         const float* __restrict__ b, float* __restrict__ scores) {
  __shared__ float red[256];
  int row = blockIdx.x, t = threadIdx.x;
  red[t] = h[(size_t)row * FD + t] * w[t]; __syncthreads();
  for (int o = 128; o > 0; o >>= 1) { if (t < o) red[t] += red[t + o]; __syncthreads(); }
  if (t == 0) scores[row] = 1.f / (1.f + expf(-(red[0] + b[0])));
}

__global__ void k_fill_pad(float* __restrict__ key, int* __restrict__ val,
                           const float* __restrict__ scores, int n) {
  int i = blockIdx.x * blockDim.x + threadIdx.x;
  if (i < NN) { key[i] = (i < n) ? scores[i] : -__builtin_huge_valf(); val[i] = i; }
}

// full 4096-element bitonic sort, descending, stable tie-break on index
__global__ __launch_bounds__(1024) void k_bitonic4096(float* __restrict__ key,
                                                      int* __restrict__ val) {
  __shared__ float k[NN];
  __shared__ int   v[NN];
  int t = threadIdx.x;
  for (int i = t; i < NN; i += 1024) { k[i] = key[i]; v[i] = val[i]; }
  __syncthreads();
  for (int sz = 2; sz <= NN; sz <<= 1) {
    for (int str = sz >> 1; str > 0; str >>= 1) {
      for (int n = t; n < NN / 2; n += 1024) {
        int i = 2 * n - (n & (str - 1));
        int j = i + str;
        bool desc = ((i & sz) == 0);
        float ki = k[i], kj = k[j]; int vi = v[i], vj = v[j];
        bool iBeforeJ = (ki > kj) || (ki == kj && vi < vj);
        if (desc ? !iBeforeJ : iBeforeJ) { k[i] = kj; k[j] = ki; v[i] = vj; v[j] = vi; }
      }
      __syncthreads();
    }
  }
  for (int i = t; i < NN; i += 1024) { key[i] = k[i]; val[i] = v[i]; }
}

__global__ void k_gather(const float* __restrict__ h, const int* __restrict__ idx,
                         const float* __restrict__ vals, float* __restrict__ out) {
  int r = blockIdx.x, t = threadIdx.x;
  out[(size_t)r * FD + t] = h[(size_t)idx[r] * FD + t] * vals[r];
}

__global__ void k_scatter(const float* __restrict__ h, const int* __restrict__ idx,
                          float* __restrict__ out) {
  int r = blockIdx.x, t = threadIdx.x;
  out[(size_t)idx[r] * FD + t] = h[(size_t)r * FD + t];
}

__global__ void k_zero(float* __restrict__ p, size_t n) {
  size_t stride = (size_t)gridDim.x * blockDim.x;
  for (size_t i = (size_t)blockIdx.x * blockDim.x + threadIdx.x; i < n; i += stride)
    p[i] = 0.f;
}

// ---------------- bit-packed 2-hop reachability with fused gather ----------
__global__ void k_packbits(const float* __restrict__ g, int cols, int ld,
                           unsigned* __restrict__ packed) {
  int row = blockIdx.x, w = threadIdx.x;      // blockDim = 128
  unsigned bits = 0; int base = w * 32;
  for (int j = 0; j < 32; ++j) {
    int c = base + j;
    if (c < cols && g[(size_t)row * ld + c] != 0.f) bits |= (1u << j);
  }
  packed[(size_t)row * 128 + w] = bits;
}

// out[a][b] = any(row[idx[a]] & row[idx[b]]) over packed words (graph symmetric)
__global__ __launch_bounds__(256) void k_reach(const unsigned* __restrict__ packed,
                                               const int* __restrict__ idx, int kk,
                                               int words, float* __restrict__ out,
                                               int ldo) {
  __shared__ unsigned Aw[32 * 128];
  __shared__ unsigned Bw[32 * 128];
  int t = threadIdx.x;
  int a0 = blockIdx.y * 32, b0 = blockIdx.x * 32;
  for (int i = t; i < 32 * 128; i += 256) {
    int r = i >> 7, w = i & 127;
    unsigned va = 0, vb = 0;
    if (w < words) {
      if (a0 + r < kk) va = packed[(size_t)idx[a0 + r] * 128 + w];
      if (b0 + r < kk) vb = packed[(size_t)idx[b0 + r] * 128 + w];
    }
    Aw[i] = va; Bw[i] = vb;
  }
  __syncthreads();
  for (int q = 0; q < 4; ++q) {
    int o = q * 256 + t;
    int ra = o >> 5, cb = o & 31;
    const unsigned* ap = &Aw[ra * 128];
    const unsigned* bp = &Bw[cb * 128];
    unsigned found = 0;
    for (int w = 0; w < words; ++w) { if (ap[w] & bp[w]) { found = 1; break; } }
    int ga = a0 + ra, gb = b0 + cb;
    if (ga < kk && gb < kk) out[(size_t)ga * ldo + gb] = (float)found;
  }
}

// ---------------- deterministic reductions ----------------
__global__ void k_sum_partial(const float* __restrict__ x, size_t n,
                              float* __restrict__ part) {
  __shared__ float red[256];
  float s = 0.f;
  size_t stride = (size_t)gridDim.x * blockDim.x;
  for (size_t i = (size_t)blockIdx.x * blockDim.x + threadIdx.x; i < n; i += stride)
    s += x[i];
  red[threadIdx.x] = s; __syncthreads();
  for (int o = 128; o > 0; o >>= 1) {
    if (threadIdx.x < o) red[threadIdx.x] += red[threadIdx.x + o];
    __syncthreads();
  }
  if (threadIdx.x == 0) part[blockIdx.x] = red[0];
}

__global__ void k_sum_final(const float* __restrict__ part, int n,
                            float* __restrict__ out, float scale) {
  __shared__ float red[256];
  float s = 0.f;
  for (int i = threadIdx.x; i < n; i += 256) s += part[i];
  red[threadIdx.x] = s; __syncthreads();
  for (int o = 128; o > 0; o >>= 1) {
    if (threadIdx.x < o) red[threadIdx.x] += red[threadIdx.x + o];
    __syncthreads();
  }
  if (threadIdx.x == 0) *out = red[0] * scale;
}

__global__ void k_bce_partial(const float* __restrict__ g, const float* __restrict__ og,
                              const float* __restrict__ onesPtr, size_t n,
                              float* __restrict__ part) {
  __shared__ float red[256];
  float ones = *onesPtr;
  float tn = (float)n;
  float w1 = tn * 0.5f / ones;
  float w0 = tn * 0.5f / (tn - ones);
  float s = 0.f;
  size_t stride = (size_t)gridDim.x * blockDim.x;
  for (size_t i = (size_t)blockIdx.x * blockDim.x + threadIdx.x; i < n; i += stride) {
    float gv = g[i];
    float o = og[i];
    o = fminf(fmaxf(o, 1e-12f), 1.f - 1e-12f);
    float w = (gv == 0.f) ? w0 : w1;
    s += w * -(gv * logf(o) + (1.f - gv) * logf(1.f - o));
  }
  red[threadIdx.x] = s; __syncthreads();
  for (int o = 128; o > 0; o >>= 1) {
    if (threadIdx.x < o) red[threadIdx.x] += red[threadIdx.x + o];
    __syncthreads();
  }
  if (threadIdx.x == 0) part[blockIdx.x] = red[0];
}

// =====================================================================
// host orchestration
// =====================================================================
// input flattening (setup_inputs dict insertion order, depth-first):
enum {
  IN_G = 0, IN_H = 1,
  IN_SGCN_W = 2, IN_SGCN_B, IN_SLN_S, IN_SLN_BI,
  IN_E0_GW = 6, IN_E0_GB, IN_E0_LS, IN_E0_LBI, IN_E0_PW, IN_E0_PB,
  IN_E1_GW = 12, IN_E1_GB, IN_E1_LS, IN_E1_LBI, IN_E1_PW, IN_E1_PB,
  IN_BOT_W = 18, IN_BOT_B, IN_BOTLN_S, IN_BOTLN_BI,
  IN_DEC1 = 22,   // 14 leaves: l0{iln.s,iln.bi,W,b,ln.s,ln.bi} l1{...} out{s,bi}
  IN_DEC2 = 36,
  IN_R1_W = 50, IN_R1_B, IN_R2_W, IN_R2_B
};

extern "C" void kernel_launch(void* const* d_in, const int* in_sizes, int n_in,
                              void* d_out, int out_size, void* d_ws, size_t ws_size,
                              hipStream_t stream) {
  (void)in_sizes; (void)n_in; (void)out_size;
  auto P = [&](int i) { return (const float*)d_in[i]; };

  // ---- workspace carve-up (float slots, 256B aligned) ----
  float* ws = (float*)d_ws;
  size_t off = 0;
  auto alloc = [&](size_t n) { float* p = ws + off; off += (n + 63) & ~(size_t)63; return p; };
  float*    G0   = alloc((size_t)NN * NN);
  float*    G1   = alloc((size_t)KK0 * KK0_LD);
  float*    G2   = alloc((size_t)KK1 * KK1_LD);
  __bf16*   G0b  = (__bf16*)alloc((size_t)NN * NN / 2);
  __bf16*   G1b  = (__bf16*)alloc((size_t)KK0 * KK0_LD / 2);
  __bf16*   G2b  = (__bf16*)alloc((size_t)KK1 * KK1_LD / 2);
  unsigned* PACK = (unsigned*)alloc((size_t)NN * 128);
  float*    T    = alloc((size_t)NN * FD);
  float*    HA   = alloc((size_t)NN * FD);
  float*    HB   = alloc((size_t)NN * FD);
  float*    ORI  = alloc((size_t)NN * FD);
  float*    DO0  = alloc((size_t)NN * FD);
  float*    DO1  = alloc((size_t)NN * FD);
  float*    HFULL= alloc((size_t)NN * FD);
  float*    HBOT = alloc((size_t)NN * FD);
  float*    OUT1 = alloc((size_t)NN * FD);
  float*    OUT2 = alloc((size_t)NN * FD);
  float*    R1   = alloc((size_t)NN * FD);
  float*    R2   = alloc((size_t)NN * FD);
  float*    DEG  = alloc(NN);
  float*    SKEY = alloc(NN);
  int*      SVAL = (int*)alloc(NN);
  int*      IDX0 = (int*)alloc(NN);
  float*    VAL0 = alloc(NN);
  int*      IDX1 = (int*)alloc(NN);
  float*    VAL1 = alloc(NN);
  float*    PART = alloc(4096);
  float*    SC   = alloc(64);
  float*    SCOR = alloc(NN);
  if (ws_size < off * sizeof(float)) return;   // workspace too small: bail safely

  const float* g  = P(IN_G);
  const float* h0 = P(IN_H);
  float* loss = (float*)d_out;
  float* o_g  = (float*)d_out + 1;
  const size_t HB_BYTES = (size_t)NN * FD * sizeof(float);

  auto gemm = [&](const float* A, const float* B, float* C, int M, int N, int K,
                  int lda, int ldb, int ldc, const float* bias, int mode, int transB) {
    dim3 grid((N + 63) / 64, (M + 63) / 64);
    k_gemm_bf16<<<grid, 256, 0, stream>>>(A, B, C, M, N, K, lda, ldb, ldc, bias, mode, transB);
  };
  auto gemmA = [&](const __bf16* Ab, const float* B, float* C, int M, int K, int lda) {
    dim3 grid((FD + 63) / 64, (M + 63) / 64);
    k_gemm_bfA<<<grid, 256, 0, stream>>>(Ab, B, C, M, FD, K, lda, FD, FD);
  };
  auto normg = [&](float* G, int n, int ld, __bf16* Gb) {  // in-place norm_g (+ fused bf16)
    k_rowdeg<<<n, 256, 0, stream>>>(G, n, ld, DEG);
    k_colscale<<<2048, 256, 0, stream>>>(G, G, Gb, DEG, (size_t)n, (size_t)n, (size_t)ld);
  };

  // ones = sum(g)  (for BCE weights)
  k_sum_partial<<<1024, 256, 0, stream>>>(g, (size_t)NN * NN, PART);
  k_sum_final<<<1, 256, 0, stream>>>(PART, 1024, SC, 1.f);

  // gn = norm_g(g) -> G0 (+ bf16 shadow)
  k_rowdeg<<<NN, 256, 0, stream>>>(g, NN, NN, DEG);
  k_colscale<<<2048, 256, 0, stream>>>(g, G0, G0b, DEG, NN, NN, NN);

  // h = LN(relu((gn@h) @ W + b))   [s_gcn, s_ln]
  gemmA(G0b, h0, T, NN, NN, NN);
  gemm(T, P(IN_SGCN_W), HA, NN, FD, FD, FD, FD, FD, P(IN_SGCN_B), 1, 0);
  k_layernorm<<<NN, 256, 0, stream>>>(HA, nullptr, P(IN_SLN_S), P(IN_SLN_BI), HA);
  hipMemcpyAsync(ORI, HA, HB_BYTES, hipMemcpyDeviceToDevice, stream);

  // ---- encoder level 0 (4096 nodes) ----
  normg(G0, NN, NN, G0b);                         // adj_ms[0] = norm_g(gn)
  gemmA(G0b, HA, T, NN, NN, NN);
  gemm(T, P(IN_E0_GW), HB, NN, FD, FD, FD, FD, FD, P(IN_E0_GB), 1, 0);
  k_layernorm<<<NN, 256, 0, stream>>>(HA, HB, P(IN_E0_LS), P(IN_E0_LBI), HA);
  hipMemcpyAsync(DO0, HA, HB_BYTES, hipMemcpyDeviceToDevice, stream);
  k_scores<<<NN, 256, 0, stream>>>(HA, P(IN_E0_PW), P(IN_E0_PB), SCOR);
  k_fill_pad<<<NN / 256, 256, 0, stream>>>(SKEY, SVAL, SCOR, NN);
  k_bitonic4096<<<1, 1024, 0, stream>>>(SKEY, SVAL);
  hipMemcpyAsync(IDX0, SVAL, NN * sizeof(int),   hipMemcpyDeviceToDevice, stream);
  hipMemcpyAsync(VAL0, SKEY, NN * sizeof(float), hipMemcpyDeviceToDevice, stream);
  k_gather<<<KK0, 256, 0, stream>>>(HA, IDX0, VAL0, HB);   // HB = pooled h (KK0)
  // 2-hop reachability on gn2 pattern, gathered -> G1
  k_packbits<<<NN, 128, 0, stream>>>(G0, NN, NN, PACK);
  {
    dim3 rg((KK0 + 31) / 32, (KK0 + 31) / 32);
    k_reach<<<rg, 256, 0, stream>>>(PACK, IDX0, KK0, NN / 32, G1, KK0_LD);
  }
  normg(G1, KK0, KK0_LD, nullptr);                // top_k_graph's norm_g

  // ---- encoder level 1 (KK0 nodes) ----
  normg(G1, KK0, KK0_LD, G1b);                    // adj_ms[1]
  gemmA(G1b, HB, T, KK0, KK0, KK0_LD);
  gemm(T, P(IN_E1_GW), HA, KK0, FD, FD, FD, FD, FD, P(IN_E1_GB), 1, 0);
  k_layernorm<<<KK0, 256, 0, stream>>>(HB, HA, P(IN_E1_LS), P(IN_E1_LBI), HB);
  hipMemcpyAsync(DO1, HB, (size_t)KK0 * FD * sizeof(float), hipMemcpyDeviceToDevice, stream);
  k_scores<<<KK0, 256, 0, stream>>>(HB, P(IN_E1_PW), P(IN_E1_PB), SCOR);
  k_fill_pad<<<NN / 256, 256, 0, stream>>>(SKEY, SVAL, SCOR, KK0);
  k_bitonic4096<<<1, 1024, 0, stream>>>(SKEY, SVAL);
  hipMemcpyAsync(IDX1, SVAL, NN * sizeof(int),   hipMemcpyDeviceToDevice, stream);
  hipMemcpyAsync(VAL1, SKEY, NN * sizeof(float), hipMemcpyDeviceToDevice, stream);
  k_gather<<<KK1, 256, 0, stream>>>(HB, IDX1, VAL1, HA);   // HA = pooled h (KK1)
  k_packbits<<<KK0, 128, 0, stream>>>(G1, KK0, KK0_LD, PACK);
  {
    dim3 rg((KK1 + 31) / 32, (KK1 + 31) / 32);
    k_reach<<<rg, 256, 0, stream>>>(PACK, IDX1, KK1, (KK0 + 31) / 32, G2, KK1_LD);
  }
  normg(G2, KK1, KK1_LD, nullptr);                // top_k_graph's norm_g
  normg(G2, KK1, KK1_LD, G2b);                    // gb = norm_g(...) in embed_one

  // ---- bottleneck ----
  gemmA(G2b, HA, T, KK1, KK1, KK1_LD);
  gemm(T, P(IN_BOT_W), HB, KK1, FD, FD, FD, FD, FD, P(IN_BOT_B), 1, 0);
  k_layernorm<<<KK1, 256, 0, stream>>>(HB, nullptr, P(IN_BOTLN_S), P(IN_BOTLN_BI), HBOT);

  // decoders re-normalize adj_ms entries (same result for both decoders)
  normg(G1, KK0, KK0_LD, G1b);
  normg(G0, NN, NN, G0b);

  auto decoder = [&](int b, float* OUT) {
    // level 0 of decoder (u=1): unpool KK1 -> KK0
    k_zero<<<256, 256, 0, stream>>>(HFULL, (size_t)KK0 * FD);
    k_scatter<<<KK1, 256, 0, stream>>>(HBOT, IDX1, HFULL);
    k_layernorm<<<KK0, 256, 0, stream>>>(DO1, HFULL, P(b + 0), P(b + 1), HFULL); // h1
    gemmA(G1b, HFULL, T, KK0, KK0, KK0_LD);
    gemm(T, P(b + 2), HA, KK0, FD, FD, FD, FD, FD, P(b + 3), 1, 0);
    k_layernorm<<<KK0, 256, 0, stream>>>(HA, HFULL, P(b + 4), P(b + 5), HA);
    // level 1 of decoder (u=0): unpool KK0 -> NN
    k_zero<<<256, 256, 0, stream>>>(HFULL, (size_t)NN * FD);
    k_scatter<<<KK0, 256, 0, stream>>>(HA, IDX0, HFULL);
    k_layernorm<<<NN, 256, 0, stream>>>(DO0, HFULL, P(b + 6), P(b + 7), HFULL);  // h1
    gemmA(G0b, HFULL, T, NN, NN, NN);
    gemm(T, P(b + 8), HA, NN, FD, FD, FD, FD, FD, P(b + 9), 1, 0);
    k_layernorm<<<NN, 256, 0, stream>>>(HA, HFULL, P(b + 10), P(b + 11), HA);
    k_layernorm<<<NN, 256, 0, stream>>>(HA, ORI, P(b + 12), P(b + 13), OUT);     // out_ln
  };
  decoder(IN_DEC1, OUT1);
  decoder(IN_DEC2, OUT2);

  // reduce1 / reduce2 (bias, no relu)
  gemm(OUT1, P(IN_R1_W), R1, NN, FD, FD, FD, FD, FD, P(IN_R1_B), 2, 0);
  gemm(OUT2, P(IN_R2_W), R2, NN, FD, FD, FD, FD, FD, P(IN_R2_B), 2, 0);

  // o_g = sigmoid(R1 @ R2^T) streamed straight into d_out
  gemm(R1, R2, o_g, NN, NN, FD, FD, FD, NN, nullptr, 3, 1);

  // weighted BCE -> loss (deterministic two-stage reduction)
  k_bce_partial<<<1024, 256, 0, stream>>>(g, o_g, SC, (size_t)NN * NN, PART);
  k_sum_final<<<1, 256, 0, stream>>>(PART, 1024, loss, 1.f / ((float)NN * (float)NN));
}
